// TransformerBlock_3856880632002
// MI455X (gfx1250) — matrix operand order repaired and checked by reading, not compiled, not run on hardware
//
#include <hip/hip_runtime.h>
#include <math.h>

// Problem constants (from reference): B=2, S=2048, D=1024, H=16, DH=64, FF=4096
#define BB 2
#define SS 2048
#define DD 1024
#define HH 16
#define DHH 64
#define FF_ 4096
#define M_TOT (BB * SS)   // 4096 token rows

typedef __attribute__((ext_vector_type(16))) __bf16 v16bf;
typedef __attribute__((ext_vector_type(8)))  float  v8f;

struct __attribute__((aligned(16))) U4 { unsigned int x, y, z, w; };

union Frag {
  v16bf          bf;
  U4             q[2];
  unsigned short us[16];
};

__device__ __forceinline__ unsigned short f2bf(float f) {
  unsigned int u = __float_as_uint(f);
  u += 0x7FFFu + ((u >> 16) & 1u);   // round-to-nearest-even
  return (unsigned short)(u >> 16);
}

__device__ __forceinline__ float red_max16(float v) {
#pragma unroll
  for (int m = 1; m < 16; m <<= 1) v = fmaxf(v, __shfl_xor(v, m, 32));
  return v;
}
__device__ __forceinline__ float red_sum16(float v) {
#pragma unroll
  for (int m = 1; m < 16; m <<= 1) v += __shfl_xor(v, m, 32);
  return v;
}

// CDNA5 async global->LDS copy (ASYNCcnt path). INST_OFFSET is added to both
// the global and the LDS address (ISA 08_async_tensor 4.4), so one address
// pair serves consecutive 16 B chunks.
__device__ __forceinline__ void async_ld_b128(unsigned ldsOff,
                                              unsigned long long gaddr) {
  asm volatile("global_load_async_to_lds_b128 %0, %1, off"
               :: "v"(ldsOff), "v"(gaddr) : "memory");
}
__device__ __forceinline__ void async_ld_b128_o16(unsigned ldsOff,
                                                  unsigned long long gaddr) {
  asm volatile("global_load_async_to_lds_b128 %0, %1, off offset:16"
               :: "v"(ldsOff), "v"(gaddr) : "memory");
}
__device__ __forceinline__ void wait_async0() {
  asm volatile("s_wait_asynccnt 0" ::: "memory");
}

// ---------------------------------------------------------------------------
// fp32 -> bf16 conversion (weights)
// ---------------------------------------------------------------------------
__global__ void cvt_bf16_kernel(const float* __restrict__ src,
                                unsigned short* __restrict__ dst, int n) {
  for (int i = blockIdx.x * blockDim.x + threadIdx.x; i < n;
       i += gridDim.x * blockDim.x)
    dst[i] = f2bf(src[i]);
}

// ---------------------------------------------------------------------------
// RMSNorm: one block per token row, fp32 math, bf16 output
// ---------------------------------------------------------------------------
__global__ __launch_bounds__(256) void rmsnorm_kernel(
    const float* __restrict__ x, const float* __restrict__ g,
    unsigned short* __restrict__ out) {
  const int row = blockIdx.x;
  const float* xr = x + (size_t)row * DD;
  float ss = 0.f;
  for (int i = threadIdx.x; i < DD; i += 256) { float v = xr[i]; ss += v * v; }
  __shared__ float red[256];
  red[threadIdx.x] = ss;
  __syncthreads();
  for (int s = 128; s > 0; s >>= 1) {
    if (threadIdx.x < s) red[threadIdx.x] += red[threadIdx.x + s];
    __syncthreads();
  }
  const float rinv = rsqrtf(red[0] * (1.0f / DD) + 1e-5f);
  unsigned short* orow = out + (size_t)row * DD;
  for (int i = threadIdx.x; i < DD; i += 256)
    orow[i] = f2bf(xr[i] * rinv * g[i]);
}

// ---------------------------------------------------------------------------
// Tiled bf16 GEMM:  C[M,N] = A[M,K] * W[N,K]^T   (both operands K-contiguous)
// Block: 256 threads (8 waves), 128x128 C tile, K-step 32.
// Double-buffered LDS fed by GLOBAL_LOAD_ASYNC_TO_LDS_B128 (ASYNCcnt):
// tile k+1 streams global->LDS while tile k feeds the WMMA pipes.
// Wave (wm in 0..1, wn in 0..3) owns a 64x32 sub-tile = 4x2 WMMA accumulators.
// EPI: 0 = store bf16; 1 = gelu then bf16; 2 = fp32 + residual, store fp32.
// ---------------------------------------------------------------------------
#define EPI_BF16 0
#define EPI_GELU 1
#define EPI_RES  2

template <int EPI>
__global__ __launch_bounds__(256) void gemm_bf16_kernel(
    const unsigned short* __restrict__ A, const unsigned short* __restrict__ Bw,
    int N, int K, const float* __restrict__ res,
    unsigned short* __restrict__ outb, float* __restrict__ outf) {
  // 48-element (96 B) row stride keeps every 16 B fragment chunk aligned.
  __shared__ __align__(16) unsigned short lA[2][128 * 48];
  __shared__ __align__(16) unsigned short lB[2][128 * 48];

  const int tid  = threadIdx.x;
  const int lane = tid & 31;
  const int wave = tid >> 5;
  const int wm   = wave >> 2;  // 0..1
  const int wn   = wave & 3;   // 0..3
  const int blockM = blockIdx.y * 128;
  const int blockN = blockIdx.x * 128;

  const v8f vzero = {0.f, 0.f, 0.f, 0.f, 0.f, 0.f, 0.f, 0.f};
  v8f acc[4][2];
#pragma unroll
  for (int i = 0; i < 4; i++)
#pragma unroll
    for (int j = 0; j < 2; j++) acc[i][j] = vzero;

  // Cooperative tile staging: 2 threads per row, 16 bf16 (32 B) each.
  const int ldrow  = tid >> 1;
  const int ldhalf = tid & 1;
  const size_t aRowBase = (size_t)(blockM + ldrow) * K + ldhalf * 16;
  const size_t bRowBase = (size_t)(blockN + ldrow) * K + ldhalf * 16;
  const unsigned ldsStageOff = (unsigned)(ldrow * 48 + ldhalf * 16);

  auto stage = [&](int buf, int k0) {
    const unsigned long long ga =
        (unsigned long long)(size_t)(A + aRowBase + k0);
    const unsigned long long gb =
        (unsigned long long)(size_t)(Bw + bRowBase + k0);
    const unsigned la = (unsigned)(size_t)(&lA[buf][ldsStageOff]);
    const unsigned lb = (unsigned)(size_t)(&lB[buf][ldsStageOff]);
    async_ld_b128(la, ga);
    async_ld_b128_o16(la, ga);
    async_ld_b128(lb, gb);
    async_ld_b128_o16(lb, gb);
  };

  const int kg8  = (lane >> 4) * 8;   // unused; A fragments also use kg16
  const int kg16 = (lane >> 4) * 16;  // B-fragment K group
  const int mrow = wm * 64 + (lane & 15);
  const int ncol = wn * 32 + (lane & 15);

  stage(0, 0);
  wait_async0();
  __syncthreads();

  int cur = 0;
  for (int k0 = 0; k0 < K; k0 += 32) {
    if (k0 + 32 < K) stage(cur ^ 1, k0 + 32);  // overlap with compute below

    // A fragments: lane m holds 16 contiguous K values at kg16
    Frag fa[4], fb[2];
#pragma unroll
    for (int mt = 0; mt < 4; mt++) {
      const unsigned short* p = &lA[cur][(mrow + mt * 16) * 48];
      fa[mt].q[0] = *(const U4*)(p + kg16);
      fa[mt].q[1] = *(const U4*)(p + kg16 + 8);
    }
    // B fragments: lane n holds 16 contiguous K values at kg16
#pragma unroll
    for (int nt = 0; nt < 2; nt++) {
      const unsigned short* p = &lB[cur][(ncol + nt * 16) * 48 + kg16];
      fb[nt].q[0] = *(const U4*)(p);
      fb[nt].q[1] = *(const U4*)(p + 8);
    }
#pragma unroll
    for (int mt = 0; mt < 4; mt++)
#pragma unroll
      for (int nt = 0; nt < 2; nt++)
        acc[mt][nt] = __builtin_amdgcn_wmma_f32_16x16x32_bf16(
            false, fa[mt].bf, false, fb[nt].bf, (short)0, acc[mt][nt], false,
            false);

    wait_async0();     // this wave's next-tile asyncs have landed
    __syncthreads();   // every wave's asyncs + this tile's reads are done
    cur ^= 1;
  }

  // Epilogue: C layout row = i + 8*(lane>>4), col = lane&15
  const int half = lane >> 4;
  const int nc   = lane & 15;
#pragma unroll
  for (int mt = 0; mt < 4; mt++) {
#pragma unroll
    for (int nt = 0; nt < 2; nt++) {
#pragma unroll
      for (int i = 0; i < 8; i++) {
        const int r = blockM + wm * 64 + mt * 16 + i + half * 8;
        const int c = blockN + wn * 32 + nt * 16 + nc;
        const size_t idx = (size_t)r * N + c;
        const float v = acc[mt][nt][i];
        if (EPI == EPI_BF16) {
          outb[idx] = f2bf(v);
        } else if (EPI == EPI_GELU) {
          outb[idx] = f2bf(0.5f * v * (1.0f + erff(v * 0.70710678118f)));
        } else {
          outf[idx] = v + res[idx];
        }
      }
    }
  }
}

// ---------------------------------------------------------------------------
// Causal flash attention. One wave per 16-query tile (8 tiles / block).
// Q,K,V,O: bf16, layout [b*S + s][h*64 + dh]. Online softmax in WMMA C layout.
// ---------------------------------------------------------------------------
__global__ __launch_bounds__(256) void attention_kernel(
    const unsigned short* __restrict__ Q, const unsigned short* __restrict__ Kk,
    const unsigned short* __restrict__ V, unsigned short* __restrict__ O) {
  __shared__ __align__(16) unsigned short ldsP[8][16 * 48];  // per-wave staging

  const int lane = threadIdx.x & 31;
  const int wave = threadIdx.x >> 5;
  const int bh = blockIdx.y;
  const int b = bh >> 4;     // H == 16
  const int h = bh & 15;
  const int q0 = (blockIdx.x * 8 + wave) * 16;
  const int half = lane >> 4;
  const int nc   = lane & 15;

  const size_t headOff = (size_t)b * SS * DD + (size_t)h * DHH;

  // Q A-fragments for dh [0,32) and [32,64)
  Frag aq0, aq1;
  {
    const size_t rb = headOff + (size_t)(q0 + nc) * DD;
    const int kg = half * 16;
    aq0.q[0] = *(const U4*)(Q + rb + kg);
    aq0.q[1] = *(const U4*)(Q + rb + kg + 8);
    aq1.q[0] = *(const U4*)(Q + rb + 32 + kg);
    aq1.q[1] = *(const U4*)(Q + rb + 40 + kg);
  }

  const v8f vzero = {0.f, 0.f, 0.f, 0.f, 0.f, 0.f, 0.f, 0.f};
  float m[8], l[8];
#pragma unroll
  for (int i = 0; i < 8; i++) { m[i] = -3.0e38f; l[i] = 0.f; }
  v8f o[4];
#pragma unroll
  for (int t = 0; t < 4; t++) o[t] = vzero;

  unsigned short* pst = &ldsP[wave][0];
  const int ktiles = (q0 + 16 + 31) >> 5;   // causal limit, 32 keys per tile

  for (int kt = 0; kt < ktiles; kt++) {
    const int kb = kt * 32;

    // Scores S = Q * K^T (16 q x 32 keys), K is B operand: lane n = key
    v8f s0 = vzero, s1 = vzero;
    {
      Frag f;
      const size_t c0 = headOff + (size_t)(kb + nc) * DD + half * 16;
      f.q[0] = *(const U4*)(Kk + c0);
      f.q[1] = *(const U4*)(Kk + c0 + 8);
      s0 = __builtin_amdgcn_wmma_f32_16x16x32_bf16(false, aq0.bf, false, f.bf,
                                                   (short)0, s0, false, false);
      f.q[0] = *(const U4*)(Kk + c0 + 32);
      f.q[1] = *(const U4*)(Kk + c0 + 40);
      s0 = __builtin_amdgcn_wmma_f32_16x16x32_bf16(false, aq1.bf, false, f.bf,
                                                   (short)0, s0, false, false);
      const size_t c1 = c0 + (size_t)16 * DD;
      f.q[0] = *(const U4*)(Kk + c1);
      f.q[1] = *(const U4*)(Kk + c1 + 8);
      s1 = __builtin_amdgcn_wmma_f32_16x16x32_bf16(false, aq0.bf, false, f.bf,
                                                   (short)0, s1, false, false);
      f.q[0] = *(const U4*)(Kk + c1 + 32);
      f.q[1] = *(const U4*)(Kk + c1 + 40);
      s1 = __builtin_amdgcn_wmma_f32_16x16x32_bf16(false, aq1.bf, false, f.bf,
                                                   (short)0, s1, false, false);
    }

    // Online softmax: C-layout row = i + 8*half matches o[] accumulators.
#pragma unroll
    for (int i = 0; i < 8; i++) {
      const int qrow = q0 + i + half * 8;
      float v0 = s0[i] * 0.125f;            // 1/sqrt(DH)
      float v1 = s1[i] * 0.125f;
      if (kb + nc > qrow)      v0 = -3.0e38f;
      if (kb + 16 + nc > qrow) v1 = -3.0e38f;
      const float rm = red_max16(fmaxf(v0, v1));
      const float mn = fmaxf(m[i], rm);
      const float sc = __expf(m[i] - mn);
      m[i] = mn;
      const float e0 = __expf(v0 - mn);
      const float e1 = __expf(v1 - mn);
      l[i] = l[i] * sc + red_sum16(e0 + e1);
#pragma unroll
      for (int t = 0; t < 4; t++) o[t][i] *= sc;
      pst[(i + half * 8) * 48 + nc]      = f2bf(e0);
      pst[(i + half * 8) * 48 + 16 + nc] = f2bf(e1);
    }

    // Intra-wave DS write->read ordering (per-wave private staging buffer).
    asm volatile("s_wait_dscnt 0" ::: "memory");

    // Re-read P in A-fragment layout, then O += P * V
    Frag pa;
    {
      const unsigned short* pr = pst + nc * 48;
      const int kg = half * 16;
      pa.q[0] = *(const U4*)(pr + kg);
      pa.q[1] = *(const U4*)(pr + kg + 8);
    }
    const int kg16 = half * 16;
#pragma unroll
    for (int t = 0; t < 4; t++) {
      Frag fv;
#pragma unroll
      for (int j = 0; j < 16; j++) {
        const int key = kb + kg16 + j;
        fv.us[j] = V[headOff + (size_t)key * DD + t * 16 + nc];
      }
      o[t] = __builtin_amdgcn_wmma_f32_16x16x32_bf16(false, pa.bf, false, fv.bf,
                                                     (short)0, o[t], false,
                                                     false);
    }
  }

#pragma unroll
  for (int i = 0; i < 8; i++) {
    const float rl = 1.0f / l[i];
    const int qrow = q0 + i + half * 8;
    const size_t base = headOff + (size_t)qrow * DD;
#pragma unroll
    for (int t = 0; t < 4; t++) O[base + t * 16 + nc] = f2bf(o[t][i] * rl);
  }
}

// ---------------------------------------------------------------------------
// Host launcher
// ---------------------------------------------------------------------------
extern "C" void kernel_launch(void* const* d_in, const int* in_sizes, int n_in,
                              void* d_out, int out_size, void* d_ws,
                              size_t ws_size, hipStream_t stream) {
  const float* x  = (const float*)d_in[0];
  const float* wq = (const float*)d_in[1];
  const float* wk = (const float*)d_in[2];
  const float* wv = (const float*)d_in[3];
  const float* wo = (const float*)d_in[4];
  const float* w1 = (const float*)d_in[5];
  const float* w2 = (const float*)d_in[6];
  const float* g1 = (const float*)d_in[7];
  const float* g2 = (const float*)d_in[8];
  (void)in_sizes; (void)n_in; (void)out_size; (void)ws_size;

  char* ws = (char*)d_ws;
  size_t off = 0;
  auto alloc = [&](size_t bytes) {
    void* p = ws + off;
    off += (bytes + 255) & ~(size_t)255;
    return p;
  };

  unsigned short* wq_b = (unsigned short*)alloc((size_t)DD * DD * 2);
  unsigned short* wk_b = (unsigned short*)alloc((size_t)DD * DD * 2);
  unsigned short* wv_b = (unsigned short*)alloc((size_t)DD * DD * 2);
  unsigned short* wo_b = (unsigned short*)alloc((size_t)DD * DD * 2);
  unsigned short* w1_b = (unsigned short*)alloc((size_t)FF_ * DD * 2);
  unsigned short* w2_b = (unsigned short*)alloc((size_t)DD * FF_ * 2);
  unsigned short* h_b  = (unsigned short*)alloc((size_t)M_TOT * DD * 2);
  unsigned short* q_b  = (unsigned short*)alloc((size_t)M_TOT * DD * 2);
  unsigned short* k_b  = (unsigned short*)alloc((size_t)M_TOT * DD * 2);
  unsigned short* v_b  = (unsigned short*)alloc((size_t)M_TOT * DD * 2);
  unsigned short* o_b  = (unsigned short*)alloc((size_t)M_TOT * DD * 2);
  float*          x2   = (float*)alloc((size_t)M_TOT * DD * 4);
  unsigned short* h2_b = (unsigned short*)alloc((size_t)M_TOT * DD * 2);
  unsigned short* f1_b = (unsigned short*)alloc((size_t)M_TOT * FF_ * 2);

  // 1) weights -> bf16
  cvt_bf16_kernel<<<512, 256, 0, stream>>>(wq, wq_b, DD * DD);
  cvt_bf16_kernel<<<512, 256, 0, stream>>>(wk, wk_b, DD * DD);
  cvt_bf16_kernel<<<512, 256, 0, stream>>>(wv, wv_b, DD * DD);
  cvt_bf16_kernel<<<512, 256, 0, stream>>>(wo, wo_b, DD * DD);
  cvt_bf16_kernel<<<1024, 256, 0, stream>>>(w1, w1_b, FF_ * DD);
  cvt_bf16_kernel<<<1024, 256, 0, stream>>>(w2, w2_b, DD * FF_);

  // 2) h = rmsnorm(x, g1)
  rmsnorm_kernel<<<M_TOT, 256, 0, stream>>>(x, g1, h_b);

  // 3) q/k/v = h @ W^T
  dim3 gD(DD / 128, M_TOT / 128);
  gemm_bf16_kernel<EPI_BF16><<<gD, 256, 0, stream>>>(h_b, wq_b, DD, DD,
                                                     nullptr, q_b, nullptr);
  gemm_bf16_kernel<EPI_BF16><<<gD, 256, 0, stream>>>(h_b, wk_b, DD, DD,
                                                     nullptr, k_b, nullptr);
  gemm_bf16_kernel<EPI_BF16><<<gD, 256, 0, stream>>>(h_b, wv_b, DD, DD,
                                                     nullptr, v_b, nullptr);

  // 4) causal attention
  attention_kernel<<<dim3(SS / 128, BB * HH), 256, 0, stream>>>(q_b, k_b, v_b,
                                                                o_b);

  // 5) x2 = o @ wo^T + x
  gemm_bf16_kernel<EPI_RES><<<gD, 256, 0, stream>>>(o_b, wo_b, DD, DD, x,
                                                    nullptr, x2);

  // 6) h2 = rmsnorm(x2, g2)
  rmsnorm_kernel<<<M_TOT, 256, 0, stream>>>(x2, g2, h2_b);

  // 7) f1 = gelu(h2 @ w1^T)
  gemm_bf16_kernel<EPI_GELU><<<dim3(FF_ / 128, M_TOT / 128), 256, 0, stream>>>(
      h2_b, w1_b, FF_, DD, nullptr, f1_b, nullptr);

  // 8) out = f1 @ w2^T + x2
  gemm_bf16_kernel<EPI_RES><<<gD, 256, 0, stream>>>(f1_b, w2_b, DD, FF_, x2,
                                                    nullptr, (float*)d_out);
}
